// MOD_35218731827241
// MI455X (gfx1250) — matrix operand order repaired and checked by reading, not compiled, not run on hardware
//
#include <hip/hip_runtime.h>
#include <hip/hip_bf16.h>

// ---------------------------------------------------------------------------
// Types for CDNA5 WMMA (gfx1250): 16x16x32 bf16 -> f32
// ---------------------------------------------------------------------------
typedef __attribute__((ext_vector_type(16))) __bf16 v16bf;
typedef __attribute__((ext_vector_type(8)))  float  v8f;
typedef __attribute__((ext_vector_type(4)))  unsigned int v4u;
typedef __attribute__((ext_vector_type(8)))  int v8i;
typedef __attribute__((ext_vector_type(4)))  int v4i;

union Frag32B { uint4 q[2]; v16bf v; };

__device__ __forceinline__ unsigned short f32_to_bf16(float f) {
    unsigned int u = __float_as_uint(f);
    unsigned int r = u + 0x7FFFu + ((u >> 16) & 1u);   // round-to-nearest-even
    return (unsigned short)(r >> 16);
}
__device__ __forceinline__ float bf16_to_f32(unsigned short h) {
    return __uint_as_float(((unsigned int)h) << 16);
}
__device__ __forceinline__ float lrelu(float x) { return x >= 0.0f ? x : 0.2f * x; }

// ---------------------------------------------------------------------------
// conv0: direct 3x3, Cin=3, Cout=64, stride 1, pad 1.  x: (8,3,256,256) f32 NCHW
// out: bf16 NHWC (8,256,256,64), bias + lrelu fused.  K=27 -> VALU is fine.
// ---------------------------------------------------------------------------
__global__ __launch_bounds__(256) void conv0_kernel(
    const float* __restrict__ x, const float* __restrict__ w,
    const float* __restrict__ bias, unsigned short* __restrict__ out)
{
    long pix = (long)blockIdx.x * 256 + threadIdx.x;       // 8*256*256 = 524288
    int b  = (int)(pix >> 16);
    int rm = (int)(pix & 65535);
    int oy = rm >> 8, ox = rm & 255;

    float acc[64];
#pragma unroll
    for (int i = 0; i < 64; ++i) acc[i] = 0.0f;

    for (int c = 0; c < 3; ++c)
        for (int ky = 0; ky < 3; ++ky)
            for (int kx = 0; kx < 3; ++kx) {
                int iy = oy - 1 + ky, ix = ox - 1 + kx;
                float v = 0.0f;
                if (iy >= 0 && iy < 256 && ix >= 0 && ix < 256)
                    v = x[(((size_t)b * 3 + c) * 256 + iy) * 256 + ix];
                int wi = c * 9 + ky * 3 + kx;
#pragma unroll
                for (int co = 0; co < 64; ++co)
                    acc[co] += v * w[co * 27 + wi];        // uniform -> s_load
            }

    unsigned short tmp[64];
#pragma unroll
    for (int co = 0; co < 64; ++co)
        tmp[co] = f32_to_bf16(lrelu(acc[co] + bias[co]));

    uint4* dst = (uint4*)(out + (size_t)pix * 64);
#pragma unroll
    for (int i = 0; i < 8; ++i) {
        uint4 q;
        q.x = (unsigned)tmp[i*8+0] | ((unsigned)tmp[i*8+1] << 16);
        q.y = (unsigned)tmp[i*8+2] | ((unsigned)tmp[i*8+3] << 16);
        q.z = (unsigned)tmp[i*8+4] | ((unsigned)tmp[i*8+5] << 16);
        q.w = (unsigned)tmp[i*8+6] | ((unsigned)tmp[i*8+7] << 16);
        dst[i] = q;
    }
}

// ---------------------------------------------------------------------------
// Weight repack: (Cout,Cin,KH,KW) f32 -> bf16 Wmat[Cout][K], K=(ky*KW+kx)*Cin+ci
// ---------------------------------------------------------------------------
__global__ void wprep_kernel(const float* __restrict__ w,
                             unsigned short* __restrict__ wmat,
                             int Cin, int KH, int KW, int tot)
{
    int i = blockIdx.x * 256 + threadIdx.x;
    if (i >= tot) return;
    int kx = i % KW; int t = i / KW;
    int ky = t % KH; t /= KH;
    int ci = t % Cin; int co = t / Cin;
    int K = Cin * KH * KW;
    wmat[(size_t)co * K + (ky * KW + kx) * Cin + ci] = f32_to_bf16(w[i]);
}

// ---------------------------------------------------------------------------
// Implicit-GEMM conv via v_wmma_f32_16x16x32_bf16.
// Block = 256 thr = 8 waves arranged 2(M)x4(N); tile = 64 Cout x 128 pixels.
// K-step = 32.  A (weight) tile staged by the Tensor Data Mover with LDS
// padding that reproduces the 80B conflict-free row stride; B (patch) tile
// staged as contiguous 64B NHWC channel runs.
// ---------------------------------------------------------------------------
__global__ __launch_bounds__(256) void conv_wmma_kernel(
    const unsigned short* __restrict__ act,
    const unsigned short* __restrict__ wmat,
    float* __restrict__ out,
    int Hin, int Win, int Cin, int Hout, int Wout, int Cout,
    int KW, int stride, int K)
{
    constexpr int AS = 40;   // A row stride in halves (80B)
    constexpr int BS = 40;   // B row stride in halves
    __shared__ __align__(16) unsigned short As[64 * AS];
    __shared__ __align__(16) unsigned short Bs[128 * BS];

    const int tid  = threadIdx.x;
    const int lane = tid & 31;
    const int wv   = tid >> 5;
    const int wm   = wv >> 2;         // M group: 2 x 32 couts
    const int wn   = wv & 3;          // N group: 4 x 32 pixels
    const int m0   = blockIdx.y * 64;
    const long pix0 = (long)blockIdx.x * 128;
    const int HW = Hout * Wout;

    // This thread's B-staging slot: pixel p, 16-half chunk koff
    const int p    = tid >> 1;
    const int koff = (tid & 1) * 16;
    const long pg  = pix0 + p;
    const int bimg = (int)(pg / HW);
    const int rm   = (int)(pg % HW);
    const int oy   = rm / Wout, ox = rm % Wout;

#if !__has_builtin(__builtin_amdgcn_tensor_load_to_lds)
    const int ar = tid >> 2;
    const int ac = (tid & 3) * 8;
#endif

    v8f acc[2][2] = {};

    for (int kb = 0; kb < K; kb += 32) {
        // ---- stage A tile (64 rows x 32 k bf16) ----
#if __has_builtin(__builtin_amdgcn_tensor_load_to_lds)
        if (wv == 0) {
            unsigned long long ga =
                (unsigned long long)(wmat + (size_t)m0 * K + kb);
            unsigned lds_base = (unsigned)(unsigned long long)(void*)As;
            // D# group 0: count=1 | lds_addr | global_addr(57b) | type=2
            v4u g0;
            g0.x = 1u;
            g0.y = lds_base;
            g0.z = (unsigned)(ga & 0xFFFFFFFFull);
            g0.w = (unsigned)((ga >> 32) & 0x01FFFFFFull) | (2u << 30);
            // D# group 1: data_size=2B, pad_enable, pad after 16 DW by 4 DW,
            // tensor_dim0=K, tensor_dim1=64, tile_dim0=32, tile_dim1=64,
            // tensor_dim0_stride=K
            v8i g1;
            g1[0] = (int)((1u << 16) | (1u << 20) | (3u << 22) | (3u << 25));
            g1[1] = (int)(((unsigned)K & 0xFFFFu) << 16);
            g1[2] = (int)((((unsigned)K >> 16) & 0xFFFFu) | (64u << 16));
            g1[3] = (int)(32u << 16);
            g1[4] = (int)64u;
            g1[5] = (int)K;
            g1[6] = 0;
            g1[7] = 0;
            v4i g2 = {0, 0, 0, 0};
            v4i g3 = {0, 0, 0, 0};
#if __clang_major__ >= 23
            v8i g4 = {0, 0, 0, 0, 0, 0, 0, 0};
            __builtin_amdgcn_tensor_load_to_lds(g0, g1, g2, g3, g4, 0);
#else
            __builtin_amdgcn_tensor_load_to_lds(g0, g1, g2, g3, 0);
#endif
        }
#else
        *(uint4*)(As + ar * AS + ac) =
            *(const uint4*)(wmat + (size_t)(m0 + ar) * K + kb + ac);
#endif

        // ---- stage B tile (128 pixels x 32 k); one spatial tap per K-step ----
        {
            int tap = kb / Cin;
            int cib = kb - tap * Cin;
            int ty = tap / KW, tx = tap - ty * KW;
            int iy = oy * stride - 1 + ty;
            int ix = ox * stride - 1 + tx;
            uint4 v0 = {0,0,0,0}, v1 = {0,0,0,0};
            if (iy >= 0 && iy < Hin && ix >= 0 && ix < Win) {
                const uint4* src = (const uint4*)(act +
                    (((size_t)bimg * Hin + iy) * Win + ix) * Cin + cib + koff);
                v0 = src[0]; v1 = src[1];
            }
            uint4* d = (uint4*)(Bs + p * BS + koff);
            d[0] = v0; d[1] = v1;
        }

#if __has_builtin(__builtin_amdgcn_tensor_load_to_lds)
        if (wv == 0) __builtin_amdgcn_s_wait_tensorcnt(0);
#endif
        __syncthreads();

        // ---- fragments in contiguous k order (k = half*16 + s) for A and B, then 4x WMMA ----
        const int half = lane >> 4;
        const int l16  = lane & 15;
        Frag32B afr[2];
#pragma unroll
        for (int ai = 0; ai < 2; ++ai) {
            const unsigned short* ap =
                As + (wm * 32 + ai * 16 + l16) * AS + half * 16;
            afr[ai].q[0] = *(const uint4*)ap;        // K = half*16 .. +7
            afr[ai].q[1] = *(const uint4*)(ap + 8);  // K = half*16+8 .. +15
        }
#pragma unroll
        for (int bj = 0; bj < 2; ++bj) {
            Frag32B bfr;
            const unsigned short* bp =
                Bs + (wn * 32 + bj * 16 + l16) * BS + half * 16;
            bfr.q[0] = *(const uint4*)bp;            // K = half*16 .. +7
            bfr.q[1] = *(const uint4*)(bp + 8);      // K = half*16+8 .. +15
#pragma unroll
            for (int ai = 0; ai < 2; ++ai)
                acc[ai][bj] = __builtin_amdgcn_wmma_f32_16x16x32_bf16(
                    false, afr[ai].v, false, bfr.v, (short)0,
                    acc[ai][bj], false, false);
        }
        __syncthreads();
    }

    // ---- epilogue: C/D layout -> f32 NHWC ----
    const int half = lane >> 4;
    const int l16  = lane & 15;
#pragma unroll
    for (int ai = 0; ai < 2; ++ai)
#pragma unroll
        for (int bj = 0; bj < 2; ++bj) {
            long pgo = pix0 + wn * 32 + bj * 16 + l16;
            float* op = out + (size_t)pgo * Cout + m0 + wm * 32 + ai * 16 + half * 8;
            float4 lo = { acc[ai][bj][0], acc[ai][bj][1], acc[ai][bj][2], acc[ai][bj][3] };
            float4 hi = { acc[ai][bj][4], acc[ai][bj][5], acc[ai][bj][6], acc[ai][bj][7] };
            ((float4*)op)[0] = lo;
            ((float4*)op)[1] = hi;
        }
}

// ---------------------------------------------------------------------------
// BatchNorm statistics: deterministic per-channel tree reduction.
// ---------------------------------------------------------------------------
__global__ __launch_bounds__(256) void bn_stats_kernel(
    const float* __restrict__ x, float* __restrict__ stats, long P, int C)
{
    int c = blockIdx.x;
    float s = 0.0f, s2 = 0.0f;
    for (long p = threadIdx.x; p < P; p += 256) {
        float v = x[p * C + c];
        s += v; s2 += v * v;
    }
    __shared__ float sh[512];
    sh[threadIdx.x] = s; sh[256 + threadIdx.x] = s2;
    __syncthreads();
    for (int st = 128; st; st >>= 1) {
        if (threadIdx.x < st) {
            sh[threadIdx.x]       += sh[threadIdx.x + st];
            sh[256 + threadIdx.x] += sh[256 + threadIdx.x + st];
        }
        __syncthreads();
    }
    if (threadIdx.x == 0) { stats[c] = sh[0]; stats[256 + c] = sh[256]; }
}

// ---------------------------------------------------------------------------
// BN apply + lrelu + bf16 repack (produces next conv's NHWC bf16 input)
// ---------------------------------------------------------------------------
__global__ __launch_bounds__(256) void bn_apply_kernel(
    const float* __restrict__ x, const float* __restrict__ stats,
    const float* __restrict__ g, const float* __restrict__ b,
    unsigned short* __restrict__ out, long n, int C, float invP)
{
    long i = (long)blockIdx.x * 256 + threadIdx.x;
    if (i >= n) return;
    int c = (int)(i % C);
    float mean = stats[c] * invP;
    float var  = stats[256 + c] * invP - mean * mean;
    float xn = (x[i] - mean) * rsqrtf(var + 1e-5f);
    out[i] = f32_to_bf16(lrelu(xn * g[c] + b[c]));
}

// ---------------------------------------------------------------------------
// Router: argmax over 12 expert logits per token (softmax is monotonic).
// ---------------------------------------------------------------------------
__global__ __launch_bounds__(256) void gate_kernel(
    const unsigned short* __restrict__ tok, const float* __restrict__ wg,
    int* __restrict__ idx)
{
    int wv = threadIdx.x >> 5, lane = threadIdx.x & 31;
    long t = (long)blockIdx.x * 8 + wv;
    float acc[12];
#pragma unroll
    for (int e = 0; e < 12; ++e) acc[e] = 0.0f;
    for (int d = lane; d < 256; d += 32) {
        float tv = bf16_to_f32(tok[t * 256 + d]);
        const float* wr = wg + d * 12;
#pragma unroll
        for (int e = 0; e < 12; ++e) acc[e] += tv * wr[e];
    }
#pragma unroll
    for (int off = 16; off; off >>= 1)
#pragma unroll
        for (int e = 0; e < 12; ++e) acc[e] += __shfl_xor(acc[e], off, 32);
    if (lane == 0) {
        int best = 0; float bv = acc[0];
#pragma unroll
        for (int e = 1; e < 12; ++e) if (acc[e] > bv) { bv = acc[e]; best = e; }
        idx[t] = best;
    }
}

// ---------------------------------------------------------------------------
// Top-1 expert: only the selected 256-column slice of body_w is evaluated.
// ---------------------------------------------------------------------------
__global__ __launch_bounds__(256) void expert_kernel(
    const unsigned short* __restrict__ tok, const int* __restrict__ idx,
    const float* __restrict__ bw, const float* __restrict__ bb,
    const float* __restrict__ ow, const float* __restrict__ w1,
    const float* __restrict__ b1, const float* __restrict__ w2,
    const float* __restrict__ b2, float* __restrict__ out)
{
    __shared__ float tk[256];
    __shared__ float ft[256];
    __shared__ float hs[32];
    long t = blockIdx.x;
    int tid = threadIdx.x;
    tk[tid] = bf16_to_f32(tok[t * 256 + tid]);
    __syncthreads();
    int e = idx[t];
    int wc = e * 256 + tid;
    float acc = bb[wc];
    for (int d = 0; d < 256; ++d) acc += tk[d] * bw[(size_t)d * 3072 + wc];
    acc = lrelu(acc);
    acc = lrelu(acc * ow[wc]);
    ft[tid] = acc;
    __syncthreads();
    if (tid < 32) {
        float h = b1[tid];
        for (int d = 0; d < 256; ++d) h += ft[d] * w1[d * 32 + tid];
        hs[tid] = lrelu(h);
    }
    __syncthreads();
    if (tid == 0) {
        float o = b2[0];
#pragma unroll
        for (int j = 0; j < 32; ++j) o += hs[j] * w2[j];
        out[t] = o;
    }
}

// ---------------------------------------------------------------------------
// Host driver
// ---------------------------------------------------------------------------
extern "C" void kernel_launch(void* const* d_in, const int* in_sizes, int n_in,
                              void* d_out, int out_size, void* d_ws, size_t ws_size,
                              hipStream_t stream) {
    (void)in_sizes; (void)n_in; (void)out_size; (void)ws_size;

    const float* x       = (const float*)d_in[0];
    const float* c0w     = (const float*)d_in[1];
    const float* c0b     = (const float*)d_in[2];
    const float* wg      = (const float*)d_in[21];
    const float* body_w  = (const float*)d_in[22];
    const float* body_b  = (const float*)d_in[23];
    const float* ortho_w = (const float*)d_in[24];
    const float* cls_w1  = (const float*)d_in[25];
    const float* cls_b1  = (const float*)d_in[26];
    const float* cls_w2  = (const float*)d_in[27];
    const float* cls_b2  = (const float*)d_in[28];

    char* ws = (char*)d_ws;
    unsigned short* actA = (unsigned short*)(ws);               // 67,108,864 B
    unsigned short* actB = (unsigned short*)(ws + 67108864);    // 33,554,432 B
    float*          cf32 = (float*)(ws + 100663296);            // 67,108,864 B
    unsigned short* wbase= (unsigned short*)(ws + 167772160);   // repacked weights
    float*          stats= (float*)(ws + 171524352);            // 512 f32
    int*            idxb = (int*)(ws + 171526400);              // 32768 int

    conv0_kernel<<<2048, 256, 0, stream>>>(x, c0w, c0b, actA);

    struct LayerDef { int wi, gi, bi, Cin, Cout, KH, KW, st, Hin, Win, Hout, Wout; };
    const LayerDef Ls[6] = {
        { 3,  4,  5,  64,  64, 4, 4, 2, 256, 256, 128, 128},
        { 6,  7,  8,  64, 128, 3, 3, 1, 128, 128, 128, 128},
        { 9, 10, 11, 128, 128, 4, 4, 2, 128, 128,  64,  64},
        {12, 13, 14, 128, 256, 3, 3, 1,  64,  64,  64,  64},
        {15, 16, 17, 256, 256, 3, 3, 1,  64,  64,  64,  64},
        {18, 19, 20, 256, 256, 3, 3, 1,  64,  64,  64,  64},
    };

    size_t woff = 0;
    unsigned short* inAct = actA;
    unsigned short* outAct = actB;
    for (int li = 0; li < 6; ++li) {
        const LayerDef& L = Ls[li];
        int K = L.Cin * L.KH * L.KW;
        int wtot = L.Cout * K;
        unsigned short* wmat = wbase + woff;
        woff += (size_t)wtot;

        wprep_kernel<<<(wtot + 255) / 256, 256, 0, stream>>>(
            (const float*)d_in[L.wi], wmat, L.Cin, L.KH, L.KW, wtot);

        long P = 8L * L.Hout * L.Wout;
        dim3 grid((unsigned)(P / 128), (unsigned)(L.Cout / 64));
        conv_wmma_kernel<<<grid, 256, 0, stream>>>(
            inAct, wmat, cf32,
            L.Hin, L.Win, L.Cin, L.Hout, L.Wout, L.Cout, L.KW, L.st, K);

        bn_stats_kernel<<<L.Cout, 256, 0, stream>>>(cf32, stats, P, L.Cout);

        long n = P * L.Cout;
        bn_apply_kernel<<<(unsigned)((n + 255) / 256), 256, 0, stream>>>(
            cf32, stats, (const float*)d_in[L.gi], (const float*)d_in[L.bi],
            outAct, n, L.Cout, 1.0f / (float)P);

        unsigned short* t2 = inAct; inAct = outAct; outAct = t2;
    }

    gate_kernel<<<4096, 256, 0, stream>>>(inAct, wg, idxb);
    expert_kernel<<<32768, 256, 0, stream>>>(
        inAct, idxb, body_w, body_b, ortho_w,
        cls_w1, cls_b1, cls_w2, cls_b2, (float*)d_out);
}
